// MultiHeadedAttention_16260746182842
// MI455X (gfx1250) — compile-verified, matrix operand order checked by reading, not run on hardware
//
#include <hip/hip_runtime.h>

typedef __attribute__((ext_vector_type(16))) __bf16 v16bf;
typedef __attribute__((ext_vector_type(8)))  __bf16 v8bf;
typedef __attribute__((ext_vector_type(8)))  float  v8f;

#define D_MODEL 2048
#define SEQ     2048
#define NH      16
#define DK      128
#define BATCH   2
#define MROWS   (BATCH * SEQ)   // 4096

// ------------------------------------------------- staging path selection ---
// Prefer TDM (tensor DMA), then per-lane async global->LDS, else manual.
#if __has_builtin(__builtin_amdgcn_tensor_load_to_lds) && \
    __has_builtin(__builtin_amdgcn_s_wait_tensorcnt)
#define GEMM_STAGE_TDM 1
#elif __has_builtin(__builtin_amdgcn_global_load_async_to_lds_b128) && \
      __has_builtin(__builtin_amdgcn_s_wait_asynccnt)
#define GEMM_STAGE_ASYNC 1
#endif

// ---------------------------------------------------------------- helpers ---

__device__ inline v8f wmma_bf16(v16bf a, v16bf b, v8f c) {
  // (neg_a, A, neg_b, B, c_mod, C, reuse_a, reuse_b)
  return __builtin_amdgcn_wmma_f32_16x16x32_bf16(false, a, false, b, (short)0, c,
                                                 false, false);
}

// A-fragment (16x32 bf16): lane holds row M=lane%16; elems 0..7 = K 8h..8h+7,
// elems 8..15 = K 16+8h..16+8h+7, h=lane/16. Two 16-byte loads.
__device__ inline v16bf load_frag(const __bf16* lo, const __bf16* hi) {
  union { v16bf v; v8bf h[2]; } u;
  u.h[0] = *(const v8bf*)lo;
  u.h[1] = *(const v8bf*)hi;
  return u.v;
}

__device__ inline float rmax16(float v) {
#pragma unroll
  for (int m = 8; m > 0; m >>= 1) v = fmaxf(v, __shfl_xor(v, m, 32));
  return v;
}
__device__ inline float rsum16(float v) {
#pragma unroll
  for (int m = 8; m > 0; m >>= 1) v += __shfl_xor(v, m, 32);
  return v;
}

#if defined(GEMM_STAGE_TDM)
typedef unsigned int tdm_u32x4 __attribute__((ext_vector_type(4)));
typedef int          tdm_i32x4 __attribute__((ext_vector_type(4)));
typedef int          tdm_i32x8 __attribute__((ext_vector_type(8)));

// DMA a 128-row x 32-col bf16 tile (row stride 2048 elems) into LDS.
// D# built per CDNA5 ISA 8.3/8.4: data_size=2B, tile_dim0=32, tile_dim1=128.
__device__ inline void tdm_load_tile_128x32(const __bf16* gsrc,
                                            unsigned lds_off) {
  unsigned long long ga = (unsigned long long)gsrc;
  tdm_u32x4 g0;
  g0[0] = 1u;                                   // count=1 (valid), user mode
  g0[1] = lds_off;                              // lds_addr (bytes)
  g0[2] = (unsigned)ga;                         // global_addr[31:0]
  g0[3] = ((unsigned)(ga >> 32) & 0x01FFFFFFu)  // global_addr[56:32]
          | 0x80000000u;                        // type=2 ("image")
  tdm_i32x8 g1;
  g1[0] = 0x00010000;                 // wg_mask=0, data_size=1 (2 bytes)
  g1[1] = (int)(2048u << 16);         // tensor_dim0[15:0] = 2048
  g1[2] = (int)(128u << 16);          // tensor_dim0 hi=0 | tensor_dim1 = 128
  g1[3] = (int)(32u << 16);           // tensor_dim1 hi=0 | tile_dim0 = 32
  g1[4] = 128;                        // tile_dim1 = 128, tile_dim2 = 0
  g1[5] = 2048;                       // tensor_dim0_stride[31:0]
  g1[6] = 0;                          // stride hi | tensor_dim1_stride lo
  g1[7] = 0;
  tdm_i32x4 gz = {0, 0, 0, 0};
#if __clang_major__ <= 22
  __builtin_amdgcn_tensor_load_to_lds(g0, g1, gz, gz, 0);
#else
  tdm_i32x8 gz8 = {0, 0, 0, 0, 0, 0, 0, 0};
  __builtin_amdgcn_tensor_load_to_lds(g0, g1, gz, gz, gz8, 0);
#endif
}
#endif  // GEMM_STAGE_TDM

#if defined(GEMM_STAGE_ASYNC)
// Per-lane 16-byte async copy global -> LDS (GLOBAL_LOAD_ASYNC_TO_LDS_B128).
__device__ inline void async_cp16(const __bf16* g, __bf16* l) {
  __builtin_amdgcn_global_load_async_to_lds_b128(
      (__attribute__((address_space(1))) void*)(unsigned long long)g,
      (__attribute__((address_space(3))) void*)(unsigned long long)l, 0, 0);
}
#endif  // GEMM_STAGE_ASYNC

// ---------------------------------------------------------- prep kernels ----

__global__ void cvt_bf16(const float* __restrict__ in, __bf16* __restrict__ out,
                         long n) {
  long i = (long)blockIdx.x * 256 + threadIdx.x;
  if (i < n) out[i] = (__bf16)in[i];
}

// out[n][k] = (bf16) in[k][n]   (2048 x 2048)
__global__ void transp_bf16(const float* __restrict__ in,
                            __bf16* __restrict__ out) {
  __shared__ float t[32][33];
  int x  = blockIdx.x * 32 + threadIdx.x;
  int y0 = blockIdx.y * 32;
  for (int i = threadIdx.y; i < 32; i += 8)
    t[i][threadIdx.x] = in[(long)(y0 + i) * D_MODEL + x];
  __syncthreads();
  int xo = y0 + threadIdx.x;
  int yo = blockIdx.x * 32;
  for (int i = threadIdx.y; i < 32; i += 8)
    out[(long)(yo + i) * D_MODEL + xo] = (__bf16)t[threadIdx.x][i];
}

__global__ void rope_tables(float* __restrict__ c, float* __restrict__ s) {
  long i = (long)blockIdx.x * 256 + threadIdx.x;
  if (i >= (long)SEQ * DK) return;
  int m = (int)(i >> 7);
  int j = (int)(i & (DK - 1));
  // t_j = exp(-(2*ln(1e4)/128) * (j>>1))
  float t = __expf(-0.14391156831212787f * (float)(j >> 1));
  float ang = (float)m * t;
  c[i] = __cosf(ang);
  s[i] = __sinf(ang);
}

// --------------------------------------------------------------- GEMM -------
// C[4096 x 2048] = A[4096 x 2048] @ W + bias, W pre-transposed (Bt rows are
// output columns, so B-fragments load with the A-fragment per-lane pattern).
// MODE 0: +RoPE, store bf16 [B,H,S,DK]      (Q and K)
// MODE 2: +RoPE, store bf16 [B,H,DK,S]      (V, transposed for the PV stage)
// MODE 3: store fp32 [B,S,D]                (output projection)
template <int MODE>
__global__ __launch_bounds__(256) void gemm_wmma(
    const __bf16* __restrict__ A, const __bf16* __restrict__ Bt,
    const float* __restrict__ bias, const float* __restrict__ cosT,
    const float* __restrict__ sinT, void* __restrict__ outp) {
  __shared__ __bf16 As[2][128 * 32];
  __shared__ __bf16 Bs[2][128 * 32];
  const int tid = threadIdx.x;
  const int lane = tid & 31, w = tid >> 5;
  const int half = lane >> 4, l16 = lane & 15;
  const int wm = (w & 3) * 32;   // 4 waves along M
  const int wn = (w >> 2) * 64;  // 2 waves along N
  const long mBase = (long)blockIdx.y * 128;
  const long nBase = (long)blockIdx.x * 128;

  const v8f zf = {0.f, 0.f, 0.f, 0.f, 0.f, 0.f, 0.f, 0.f};
  v8f acc[2][4];
#pragma unroll
  for (int i = 0; i < 2; ++i)
#pragma unroll
    for (int j = 0; j < 4; ++j) acc[i][j] = zf;

  const int ldRow = tid >> 1;         // 0..127
  const int ldCol = (tid & 1) * 16;   // 0 or 16
  const __bf16* gA = A  + (mBase + ldRow) * D_MODEL + ldCol;  // per-thread
  const __bf16* gB = Bt + (nBase + ldRow) * D_MODEL + ldCol;
  const __bf16* Abase = A  + mBase * D_MODEL;                 // tile base
  const __bf16* Bbase = Bt + nBase * D_MODEL;
  (void)Abase; (void)Bbase; (void)gA; (void)gB;

  int buf = 0;
  // ---- prologue: stage first K-tile into buffer 0 ----
#if defined(GEMM_STAGE_TDM)
  if (tid == 0) {
    tdm_load_tile_128x32(Abase, (unsigned)(unsigned long long)&As[0][0]);
    tdm_load_tile_128x32(Bbase, (unsigned)(unsigned long long)&Bs[0][0]);
  }
#elif defined(GEMM_STAGE_ASYNC)
  async_cp16(gA,     &As[0][ldRow * 32 + ldCol]);
  async_cp16(gA + 8, &As[0][ldRow * 32 + ldCol + 8]);
  async_cp16(gB,     &Bs[0][ldRow * 32 + ldCol]);
  async_cp16(gB + 8, &Bs[0][ldRow * 32 + ldCol + 8]);
#endif

  for (int k0 = 0; k0 < D_MODEL; k0 += 32) {
    const int nxt = k0 + 32;
#if defined(GEMM_STAGE_TDM)
    // Wave 0 issues next tile's DMA, then drains the current tile's 2 ops.
    if (tid == 0) {
      if (nxt < D_MODEL) {
        tdm_load_tile_128x32(Abase + nxt,
                             (unsigned)(unsigned long long)&As[buf ^ 1][0]);
        tdm_load_tile_128x32(Bbase + nxt,
                             (unsigned)(unsigned long long)&Bs[buf ^ 1][0]);
        __builtin_amdgcn_s_wait_tensorcnt(2);
      } else {
        __builtin_amdgcn_s_wait_tensorcnt(0);
      }
    }
    __syncthreads();
#elif defined(GEMM_STAGE_ASYNC)
    if (nxt < D_MODEL) {
      async_cp16(gA + nxt,     &As[buf ^ 1][ldRow * 32 + ldCol]);
      async_cp16(gA + nxt + 8, &As[buf ^ 1][ldRow * 32 + ldCol + 8]);
      async_cp16(gB + nxt,     &Bs[buf ^ 1][ldRow * 32 + ldCol]);
      async_cp16(gB + nxt + 8, &Bs[buf ^ 1][ldRow * 32 + ldCol + 8]);
      __builtin_amdgcn_s_wait_asynccnt(4);
    } else {
      __builtin_amdgcn_s_wait_asynccnt(0);
    }
    __syncthreads();
#else
    // Manual staging: global -> VGPR -> LDS, single buffer.
    __builtin_prefetch(gA + k0 + 32, 0, 1);
    __builtin_prefetch(gB + k0 + 32, 0, 1);
    {
      v8bf a0 = *(const v8bf*)(gA + k0);
      v8bf a1 = *(const v8bf*)(gA + k0 + 8);
      v8bf b0 = *(const v8bf*)(gB + k0);
      v8bf b1 = *(const v8bf*)(gB + k0 + 8);
      *(v8bf*)&As[0][ldRow * 32 + ldCol]     = a0;
      *(v8bf*)&As[0][ldRow * 32 + ldCol + 8] = a1;
      *(v8bf*)&Bs[0][ldRow * 32 + ldCol]     = b0;
      *(v8bf*)&Bs[0][ldRow * 32 + ldCol + 8] = b1;
    }
    __syncthreads();
#endif

    const __bf16* Ab = &As[buf][0];
    const __bf16* Bb = &Bs[buf][0];
    v16bf af[2], bfg[4];
#pragma unroll
    for (int mt = 0; mt < 2; ++mt) {
      int r = wm + mt * 16 + l16;
      af[mt] = load_frag(&Ab[r * 32 + 8 * half], &Ab[r * 32 + 16 + 8 * half]);
    }
#pragma unroll
    for (int nt = 0; nt < 4; ++nt) {
      int r = wn + nt * 16 + l16;
      bfg[nt] = load_frag(&Bb[r * 32 + 8 * half], &Bb[r * 32 + 16 + 8 * half]);
    }
#pragma unroll
    for (int mt = 0; mt < 2; ++mt)
#pragma unroll
      for (int nt = 0; nt < 4; ++nt)
        acc[mt][nt] = wmma_bf16(af[mt], bfg[nt], acc[mt][nt]);
    __syncthreads();
#if defined(GEMM_STAGE_TDM) || defined(GEMM_STAGE_ASYNC)
    buf ^= 1;
#endif
  }

  // Epilogue. C layout: VGPR r -> row r + 8*half, col = l16.
#pragma unroll
  for (int mt = 0; mt < 2; ++mt)
#pragma unroll
    for (int nt = 0; nt < 4; ++nt)
#pragma unroll
      for (int r = 0; r < 8; ++r) {
        long R = mBase + wm + mt * 16 + r + 8 * half;
        long C = nBase + wn + nt * 16 + l16;
        float v = acc[mt][nt][r] + bias[C];
        if (MODE == 3) {
          ((float*)outp)[R * D_MODEL + C] = v;
        } else {
          // RoPE: adjacent column lives in the adjacent lane (same row/half).
          float p = __shfl_xor(v, 1, 32);
          int j = (int)(C & (DK - 1));
          float rot = (j & 1) ? p : -p;  // even: -x[j+1], odd: +x[j-1]
          long s  = R & (SEQ - 1);
          long bb = R >> 11;
          float cv = cosT[s * DK + j], sv = sinT[s * DK + j];
          float o = v * cv + rot * sv;
          long hh = (C >> 7) & (NH - 1);
          __bf16* ob = (__bf16*)outp;
          if (MODE == 2)
            ob[((bb * NH + hh) * DK + j) * SEQ + s] = (__bf16)o;  // V^T
          else
            ob[((bb * NH + hh) * SEQ + s) * DK + j] = (__bf16)o;  // Q/K
        }
      }
}

// ----------------------------------------------------------- attention ------
// grid (S/128, H, B), 8 waves; wave owns 16 query rows, flash-style online
// softmax over 32 keys per iteration.
__global__ __launch_bounds__(256) void attn_wmma(
    const __bf16* __restrict__ qh, const __bf16* __restrict__ kh,
    const __bf16* __restrict__ vt, __bf16* __restrict__ outA) {
  __shared__ __bf16 Pl[8 * 16 * 32];  // per-wave 16x32 prob tile
  const int tid = threadIdx.x, lane = tid & 31, w = tid >> 5;
  const int half = lane >> 4, l16 = lane & 15;
  const int bb = blockIdx.z, hh = blockIdx.y;
  const int q0 = blockIdx.x * 128 + w * 16;
  const __bf16* Q = qh + ((long)(bb * NH + hh)) * SEQ * DK;
  const __bf16* K = kh + ((long)(bb * NH + hh)) * SEQ * DK;
  const __bf16* V = vt + ((long)(bb * NH + hh)) * DK * SEQ;

  v16bf qf[4];
  const __bf16* qrow = Q + (long)(q0 + l16) * DK;
#pragma unroll
  for (int kc = 0; kc < 4; ++kc)
    qf[kc] = load_frag(qrow + kc * 32 + 8 * half,
                       qrow + kc * 32 + 16 + 8 * half);

  const v8f zf = {0.f, 0.f, 0.f, 0.f, 0.f, 0.f, 0.f, 0.f};
  v8f O[8];
  float mi[8], li[8];
#pragma unroll
  for (int nt = 0; nt < 8; ++nt) O[nt] = zf;
#pragma unroll
  for (int r = 0; r < 8; ++r) { mi[r] = -1e30f; li[r] = 0.f; }

  __bf16* Pw = &Pl[w * 512];
  const float scale = 0.08838834764831845f;  // 1/sqrt(128)

  for (int kv = 0; kv < SEQ; kv += 32) {
    v8f s0 = zf, s1 = zf;
#pragma unroll
    for (int kc = 0; kc < 4; ++kc) {
      const __bf16* kr = K + (long)(kv + l16) * DK + kc * 32;
      v16bf kf = load_frag(kr + 8 * half, kr + 16 + 8 * half);
      s0 = wmma_bf16(qf[kc], kf, s0);
    }
#pragma unroll
    for (int kc = 0; kc < 4; ++kc) {
      const __bf16* kr = K + (long)(kv + 16 + l16) * DK + kc * 32;
      v16bf kf = load_frag(kr + 8 * half, kr + 16 + 8 * half);
      s1 = wmma_bf16(qf[kc], kf, s1);
    }

    float alpha[8];
#pragma unroll
    for (int r = 0; r < 8; ++r) {
      float a = s0[r] * scale, c = s1[r] * scale;
      float tm = rmax16(fmaxf(a, c));
      float mn = fmaxf(mi[r], tm);
      alpha[r] = __expf(mi[r] - mn);
      mi[r] = mn;
      float p0 = __expf(a - mn), p1 = __expf(c - mn);
      s0[r] = p0; s1[r] = p1;
      li[r] = li[r] * alpha[r] + rsum16(p0 + p1);
    }
#pragma unroll
    for (int nt = 0; nt < 8; ++nt)
#pragma unroll
      for (int r = 0; r < 8; ++r) O[nt][r] *= alpha[r];

    // C-layout -> A-fragment layout via per-wave LDS bounce
#pragma unroll
    for (int r = 0; r < 8; ++r) {
      int row = r + 8 * half;
      Pw[row * 32 + l16]      = (__bf16)s0[r];
      Pw[row * 32 + 16 + l16] = (__bf16)s1[r];
    }
    __syncthreads();
    v16bf pf = load_frag(&Pw[l16 * 32 + 8 * half],
                         &Pw[l16 * 32 + 16 + 8 * half]);
#pragma unroll
    for (int nt = 0; nt < 8; ++nt) {
      const __bf16* vr = V + (long)(nt * 16 + l16) * SEQ + kv;  // V^T rows
      v16bf vf = load_frag(vr + 8 * half, vr + 16 + 8 * half);
      O[nt] = wmma_bf16(pf, vf, O[nt]);
    }
    __syncthreads();
  }

#pragma unroll
  for (int nt = 0; nt < 8; ++nt)
#pragma unroll
    for (int r = 0; r < 8; ++r) {
      int srow = q0 + r + 8 * half;
      int j = nt * 16 + l16;
      float o = O[nt][r] / li[r];
      outA[((long)(bb * SEQ + srow)) * D_MODEL + hh * DK + j] = (__bf16)o;
    }
}

// ------------------------------------------------------------- launcher -----

extern "C" void kernel_launch(void* const* d_in, const int* in_sizes, int n_in,
                              void* d_out, int out_size, void* d_ws,
                              size_t ws_size, hipStream_t stream) {
  const float* query  = (const float*)d_in[0];
  const float* keys   = (const float*)d_in[1];
  const float* values = (const float*)d_in[2];
  const float* Wq = (const float*)d_in[3]; const float* bq = (const float*)d_in[4];
  const float* Wk = (const float*)d_in[5]; const float* bk = (const float*)d_in[6];
  const float* Wv = (const float*)d_in[7]; const float* bv = (const float*)d_in[8];
  const float* Wo = (const float*)d_in[9]; const float* bo = (const float*)d_in[10];

  char* ws = (char*)d_ws;
  size_t off = 0;
  auto take = [&](size_t bytes) {
    char* p = ws + off;
    off += (bytes + 255) & ~(size_t)255;
    return p;
  };
  const size_t XB = (size_t)MROWS * D_MODEL * 2;        // 16 MB
  const size_t WB = (size_t)D_MODEL * D_MODEL * 2;      // 8 MB
  const size_t HB = (size_t)BATCH * NH * SEQ * DK * 2;  // 16 MB
  const size_t TB = (size_t)SEQ * DK * 4;               // 1 MB

  __bf16* Xq  = (__bf16*)take(XB);
  __bf16* Xk  = (__bf16*)take(XB);
  __bf16* Xv  = (__bf16*)take(XB);
  __bf16* WqT = (__bf16*)take(WB);
  __bf16* WkT = (__bf16*)take(WB);
  __bf16* WvT = (__bf16*)take(WB);
  __bf16* WoT = (__bf16*)take(WB);
  __bf16* Qh  = (__bf16*)take(HB);
  __bf16* Kh  = (__bf16*)take(HB);
  __bf16* Vt  = (__bf16*)take(HB);
  __bf16* AO  = (__bf16*)take(XB);
  float* cosT = (float*)take(TB);
  float* sinT = (float*)take(TB);

  long nX = (long)MROWS * D_MODEL;
  cvt_bf16<<<(nX + 255) / 256, 256, 0, stream>>>(query, Xq, nX);
  cvt_bf16<<<(nX + 255) / 256, 256, 0, stream>>>(keys, Xk, nX);
  cvt_bf16<<<(nX + 255) / 256, 256, 0, stream>>>(values, Xv, nX);

  dim3 tg(64, 64), tb(32, 8);
  transp_bf16<<<tg, tb, 0, stream>>>(Wq, WqT);
  transp_bf16<<<tg, tb, 0, stream>>>(Wk, WkT);
  transp_bf16<<<tg, tb, 0, stream>>>(Wv, WvT);
  transp_bf16<<<tg, tb, 0, stream>>>(Wo, WoT);

  long nT = (long)SEQ * DK;
  rope_tables<<<(nT + 255) / 256, 256, 0, stream>>>(cosT, sinT);

  dim3 gg(D_MODEL / 128, MROWS / 128);
  gemm_wmma<0><<<gg, 256, 0, stream>>>(Xq, WqT, bq, cosT, sinT, Qh);
  gemm_wmma<0><<<gg, 256, 0, stream>>>(Xk, WkT, bk, cosT, sinT, Kh);
  gemm_wmma<2><<<gg, 256, 0, stream>>>(Xv, WvT, bv, cosT, sinT, Vt);

  attn_wmma<<<dim3(SEQ / 128, NH, BATCH), 256, 0, stream>>>(Qh, Kh, Vt, AO);

  gemm_wmma<3><<<gg, 256, 0, stream>>>(AO, WoT, bo, nullptr, nullptr, d_out);

  (void)in_sizes; (void)n_in; (void)out_size; (void)ws_size;
}
